// DiffNet_11519102288272
// MI455X (gfx1250) — matrix operand order repaired and checked by reading, not compiled, not run on hardware
//
#include <hip/hip_runtime.h>

// ---------------------------------------------------------------------------
// DiffNet (2-layer LSTM encoder/decoder) for MI455X / gfx1250.
// GEMMs via v_wmma_f32_16x16x32_bf16 (bf16 in, fp32 accumulate).
// Register-blocked: each wave owns a 16 x (16*NT) output strip (NT compile-
// time, no divergent guards), one A-fragment load feeds NT WMMAs.
// All fragment addresses are pointer-increment streams; weight prefetch is
// unconditional/speculative (ISA §10.5: invalid prefetch silently dropped).
// ---------------------------------------------------------------------------

typedef __bf16 bf16_t;
typedef bf16_t v16bf __attribute__((ext_vector_type(16)));
typedef float  v8f   __attribute__((ext_vector_type(8)));

union Frag { unsigned int u[8]; v16bf v; };

__device__ __forceinline__ unsigned short f2bf(float x) {
  unsigned int u = __float_as_uint(x);
  u += 0x7FFFu + ((u >> 16) & 1u);           // round to nearest even
  return (unsigned short)(u >> 16);
}

__device__ __forceinline__ float sigm(float x) {
  return 1.0f / (1.0f + __expf(-x));
}

// ---------------------------------------------------------------------------
// Accumulate acc[NT] += X(16 rows @row0, K cols) * Wp(K x 16*NT tile strip).
// A fragment ISA layout (CDNA5 ISA §7.12.2, 16-bit A 16x32):
//   lanes 0-15 : row m, VGPR0..3 = K 0..7,  VGPR4..7 = K 16..23
//   lanes 16-31: row m, VGPR0..3 = K 8..15, VGPR4..7 = K 24..31
// B fragments pre-swizzled: per (n_tile,k_tile) each lane owns 8 consecutive
// dwords (32 B -> 2x global_load_b128); consecutive k tiles are +256 dwords.
// ---------------------------------------------------------------------------
template <int NT>
__device__ __forceinline__ void gemm_src(v8f (&acc)[NT],
                                         const unsigned int* __restrict__ X,
                                         int K,
                                         const unsigned int* __restrict__ Wp,
                                         int row0, int nt0, int lane) {
  const int Kt = K >> 5;
  const unsigned int* ap =
      X + (size_t)(row0 + (lane & 15)) * (K >> 1) + ((lane >= 16) ? 4 : 0);
  const unsigned int* bp[NT];
#pragma unroll
  for (int j = 0; j < NT; ++j)
    bp[j] = Wp + ((((size_t)(nt0 + j) * Kt) * 32 + lane) << 3);

  for (int kt = 0; kt < Kt; ++kt) {
    Frag fa;
#pragma unroll
    for (int v = 0; v < 4; ++v) { fa.u[v] = ap[v]; fa.u[4 + v] = ap[8 + v]; }
    ap += 16;                                   // next 32 K-elements
#pragma unroll
    for (int j = 0; j < NT; ++j) {
      __builtin_prefetch((const void*)(bp[j] + 256), 0, 1);  // next k-tile
      Frag fb;
      const uint4* p4 = (const uint4*)bp[j];
      uint4 lo = p4[0], hi = p4[1];
      fb.u[0] = lo.x; fb.u[1] = lo.y; fb.u[2] = lo.z; fb.u[3] = lo.w;
      fb.u[4] = hi.x; fb.u[5] = hi.y; fb.u[6] = hi.z; fb.u[7] = hi.w;
      acc[j] = __builtin_amdgcn_wmma_f32_16x16x32_bf16(
          false, fa.v, false, fb.v, (short)0, acc[j], false, false);
      bp[j] += 256;
    }
  }
}

// ---------------------------------------------------------------------------
// Generic WMMA GEMM: Z[M=256][N] = X@Wp1 (+ H@Wp2) (+bias) (+add_mat),
// optional tanh, optional fp32/bf16 outputs.
// 8 waves/block; wave = 16 rows x (16*NT) cols. N == grid.x * 4 * 16 * NT.
// ---------------------------------------------------------------------------
template <int NT>
__global__ void wmma_gemm(const unsigned int* __restrict__ X, int K1,
                          const unsigned int* __restrict__ Wp1,
                          const unsigned int* __restrict__ Hm, int K2,
                          const unsigned int* __restrict__ Wp2,
                          const float* __restrict__ bias,
                          const float* __restrict__ add_mat,
                          float* __restrict__ out_f32,
                          float* __restrict__ out_f32b,
                          unsigned short* __restrict__ out_bf16,
                          int ld_bf, int col_off,
                          int N, int act) {
  const int lane = threadIdx.x & 31;
  const int wave = threadIdx.x >> 5;
  const int row0 = blockIdx.y * 32 + (wave >> 2) * 16;
  const int nt0  = (blockIdx.x * 4 + (wave & 3)) * NT;   // first n-tile

  v8f acc[NT];
#pragma unroll
  for (int j = 0; j < NT; ++j) acc[j] = v8f{};

  gemm_src<NT>(acc, X, K1, Wp1, row0, nt0, lane);
  if (K2 > 0) gemm_src<NT>(acc, Hm, K2, Wp2, row0, nt0, lane);

  // C/D layout: VGPR r -> row (row0 + r + (lane>=16)*8), col (tile + lane%16)
  const int rbase = row0 + ((lane >> 4) << 3);
#pragma unroll
  for (int j = 0; j < NT; ++j) {
    const int n = (nt0 + j) * 16 + (lane & 15);
    const float bv = bias ? bias[n] : 0.0f;
#pragma unroll
    for (int r = 0; r < 8; ++r) {
      const int m = rbase + r;
      float v = acc[j][r] + bv;
      if (add_mat) v += add_mat[(size_t)m * N + n];
      if (act) v = tanhf(v);
      if (out_f32)  out_f32 [(size_t)m * N + n] = v;
      if (out_f32b) out_f32b[(size_t)m * N + n] = v;
      if (out_bf16) out_bf16[(size_t)m * ld_bf + col_off + n] = f2bf(v);
    }
  }
}

// ---------------------------------------------------------------------------
// LSTM cell epilogue: z (B x 4H, gate order i,f,g,o) -> c (fp32), h (bf16)
// ---------------------------------------------------------------------------
__global__ void lstm_cell(const float* __restrict__ z,
                          float* __restrict__ c,
                          unsigned short* __restrict__ h_bf) {
  const int idx = blockIdx.x * blockDim.x + threadIdx.x;  // B*H = 262144
  if (idx >= 256 * 1024) return;
  const int b = idx >> 10;
  const int j = idx & 1023;
  const float* zr = z + (size_t)b * 4096;
  const float gi = sigm(zr[j]);
  const float gf = sigm(zr[j + 1024]);
  const float gg = tanhf(zr[j + 2048]);
  const float go = sigm(zr[j + 3072]);
  const float cn = gf * c[idx] + gi * gg;
  c[idx] = cn;
  h_bf[idx] = f2bf(go * tanhf(cn));
}

// ---------------------------------------------------------------------------
// Pack fp32 weight W (N x K, row-major) into WMMA B-fragment dword layout.
// One thread per bf16 pair (even k).
// ---------------------------------------------------------------------------
__global__ void pack_weight(const float* __restrict__ W,
                            unsigned int* __restrict__ out, int N, int K) {
  const int idx = blockIdx.x * blockDim.x + threadIdx.x;
  const int half = K >> 1;
  if (idx >= N * half) return;
  const int n = idx / half;
  const int k = (idx - n * half) << 1;
  const unsigned int lo = f2bf(W[(size_t)n * K + k]);
  const unsigned int hi = f2bf(W[(size_t)n * K + k + 1]);
  const int n_tile = n >> 4;
  const int k_tile = k >> 5;
  const int lane = (n & 15) + (((k & 15) >= 8) ? 16 : 0);
  const int v = ((k & 7) >> 1) + ((k & 16) >> 2);
  out[(((((size_t)n_tile * (K >> 5)) + k_tile) * 32 + lane) << 3) + v] =
      lo | (hi << 16);
}

__global__ void add_bias(const float* __restrict__ a, const float* __restrict__ b,
                         float* __restrict__ o, int n) {
  const int i = blockIdx.x * blockDim.x + threadIdx.x;
  if (i < n) o[i] = a[i] + b[i];
}

// xy = bf16(concat(pre_x, pre_y)) : (64, 256, 256)
__global__ void build_xy(const float* __restrict__ px, const float* __restrict__ py,
                         unsigned short* __restrict__ xy) {
  const int idx = blockIdx.x * blockDim.x + threadIdx.x;
  if (idx >= 64 * 256 * 256) return;
  const int t  = idx >> 16;
  const int r  = idx & 65535;
  const int bi = r >> 8;
  const int cc = r & 255;
  float v = (cc < 192) ? px[((size_t)t * 256 + bi) * 192 + cc]
                       : py[((size_t)t * 256 + bi) * 64 + (cc - 192)];
  xy[idx] = f2bf(v);
}

// xin[t][:, 0:192] = bf16(forward_x[t]); est columns written per decode step
__global__ void build_xin(const float* __restrict__ fx,
                          unsigned short* __restrict__ xin) {
  const int idx = blockIdx.x * blockDim.x + threadIdx.x;
  if (idx >= 48 * 256 * 192) return;
  const int t  = idx / (256 * 192);
  const int r  = idx - t * (256 * 192);
  const int bi = r / 192;
  const int cc = r - bi * 192;
  xin[(size_t)t * 65536 + (size_t)bi * 256 + cc] = f2bf(fx[idx]);
}

// h0=h1=0 (bf16), c0=c1=0, est = pre_y[-1]
__global__ void init_state(unsigned short* h0, unsigned short* h1,
                           float* c0, float* c1,
                           float* est, const float* __restrict__ pre_y) {
  const int idx = blockIdx.x * blockDim.x + threadIdx.x;
  if (idx >= 256 * 1024) return;
  h0[idx] = 0; h1[idx] = 0;
  c0[idx] = 0.0f; c1[idx] = 0.0f;
  if (idx < 256 * 64) est[idx] = pre_y[(size_t)63 * 256 * 64 + idx];
}

extern "C" void kernel_launch(void* const* d_in, const int* in_sizes, int n_in,
                              void* d_out, int out_size, void* d_ws, size_t ws_size,
                              hipStream_t stream) {
  const float* pre_x  = (const float*)d_in[0];
  const float* pre_y  = (const float*)d_in[1];
  const float* fwd_x  = (const float*)d_in[2];
  const float* w_ih_0 = (const float*)d_in[3];
  const float* w_hh_0 = (const float*)d_in[4];
  const float* b_ih_0 = (const float*)d_in[5];
  const float* b_hh_0 = (const float*)d_in[6];
  const float* w_ih_1 = (const float*)d_in[7];
  const float* w_hh_1 = (const float*)d_in[8];
  const float* b_ih_1 = (const float*)d_in[9];
  const float* b_hh_1 = (const float*)d_in[10];
  const float* fc_w1  = (const float*)d_in[11];
  const float* fc_b1  = (const float*)d_in[12];
  const float* fc_w2  = (const float*)d_in[13];
  const float* fc_b2  = (const float*)d_in[14];
  float* out = (float*)d_out;

  // ---- workspace carve-up (256B aligned slabs) ----
  size_t off = 0;
  auto alloc = [&](size_t bytes) -> void* {
    void* p = (char*)d_ws + off;
    off = (off + bytes + 255) & ~(size_t)255;
    return p;
  };
  unsigned int*  Wih0_p = (unsigned int*)alloc((size_t)4096 * 256  * 2);
  unsigned int*  Whh0_p = (unsigned int*)alloc((size_t)4096 * 1024 * 2);
  unsigned int*  Wih1_p = (unsigned int*)alloc((size_t)4096 * 1024 * 2);
  unsigned int*  Whh1_p = (unsigned int*)alloc((size_t)4096 * 1024 * 2);
  unsigned int*  Wfc1_p = (unsigned int*)alloc((size_t)1024 * 1024 * 2);
  unsigned int*  Wfc2_p = (unsigned int*)alloc((size_t)64   * 1024 * 2);
  float*         bias0  = (float*)alloc(4096 * 4);
  float*         bias1  = (float*)alloc(4096 * 4);
  unsigned short* xy    = (unsigned short*)alloc((size_t)64 * 256 * 256 * 2);
  unsigned short* xin   = (unsigned short*)alloc((size_t)48 * 256 * 256 * 2);
  float*         z      = (float*)alloc((size_t)256 * 4096 * 4);
  unsigned short* h0    = (unsigned short*)alloc((size_t)256 * 1024 * 2);
  unsigned short* h1    = (unsigned short*)alloc((size_t)256 * 1024 * 2);
  float*         c0     = (float*)alloc((size_t)256 * 1024 * 4);
  float*         c1     = (float*)alloc((size_t)256 * 1024 * 4);
  unsigned short* u     = (unsigned short*)alloc((size_t)256 * 1024 * 2);
  float*         est    = (float*)alloc((size_t)256 * 64 * 4);
  (void)ws_size; (void)in_sizes; (void)n_in; (void)out_size;

  const dim3 blk(256);
  auto grid1 = [](int n) { return dim3((unsigned)((n + 255) / 256)); };

  // ---- one-time prep (re-done every call; deterministic) ----
  pack_weight<<<grid1(4096 * 128),  blk, 0, stream>>>(w_ih_0, Wih0_p, 4096, 256);
  pack_weight<<<grid1(4096 * 512),  blk, 0, stream>>>(w_hh_0, Whh0_p, 4096, 1024);
  pack_weight<<<grid1(4096 * 512),  blk, 0, stream>>>(w_ih_1, Wih1_p, 4096, 1024);
  pack_weight<<<grid1(4096 * 512),  blk, 0, stream>>>(w_hh_1, Whh1_p, 4096, 1024);
  pack_weight<<<grid1(1024 * 512),  blk, 0, stream>>>(fc_w1,  Wfc1_p, 1024, 1024);
  pack_weight<<<grid1(64 * 512),    blk, 0, stream>>>(fc_w2,  Wfc2_p, 64,   1024);
  add_bias<<<grid1(4096), blk, 0, stream>>>(b_ih_0, b_hh_0, bias0, 4096);
  add_bias<<<grid1(4096), blk, 0, stream>>>(b_ih_1, b_hh_1, bias1, 4096);
  build_xy<<<grid1(64 * 256 * 256), blk, 0, stream>>>(pre_x, pre_y, xy);
  build_xin<<<grid1(48 * 256 * 192), blk, 0, stream>>>(fwd_x, xin);
  init_state<<<grid1(256 * 1024), blk, 0, stream>>>(h0, h1, c0, c1, est, pre_y);

  const dim3 gGate(16, 8);   // N=4096 = 16 blk * 4 waves * 4 tiles * 16
  const dim3 gFc1(4, 8);     // N=1024 =  4 blk * 4 waves * 4 tiles * 16
  const dim3 gFc2(1, 8);     // N=64   =  1 blk * 4 waves * 1 tile  * 16
  const dim3 gCell((256 * 1024) / 256);

  // ---- encoder: 64 steps over xy ----
  for (int t = 0; t < 64; ++t) {
    const unsigned int* xt = (const unsigned int*)(xy + (size_t)t * 256 * 256);
    wmma_gemm<4><<<gGate, blk, 0, stream>>>(xt, 256, Wih0_p,
                                            (const unsigned int*)h0, 1024, Whh0_p,
                                            bias0, nullptr, z, nullptr,
                                            nullptr, 0, 0, 4096, 0);
    lstm_cell<<<gCell, blk, 0, stream>>>(z, c0, h0);
    wmma_gemm<4><<<gGate, blk, 0, stream>>>((const unsigned int*)h0, 1024, Wih1_p,
                                            (const unsigned int*)h1, 1024, Whh1_p,
                                            bias1, nullptr, z, nullptr,
                                            nullptr, 0, 0, 4096, 0);
    lstm_cell<<<gCell, blk, 0, stream>>>(z, c1, h1);
  }

  // ---- autoregressive decoder: 48 steps ----
  for (int t = 0; t < 48; ++t) {
    // u = tanh(h1 @ fc_w1^T + fc_b1)  -> bf16
    wmma_gemm<4><<<gFc1, blk, 0, stream>>>((const unsigned int*)h1, 1024, Wfc1_p,
                                           nullptr, 0, nullptr,
                                           fc_b1, nullptr, nullptr, nullptr,
                                           u, 1024, 0, 1024, 1);
    // est = u @ fc_w2^T + fc_b2 + last_y; write est, d_out[t], xin[t][:,192:]
    unsigned short* xin_t = xin + (size_t)t * 256 * 256;
    wmma_gemm<1><<<gFc2, blk, 0, stream>>>((const unsigned int*)u, 1024, Wfc2_p,
                                           nullptr, 0, nullptr,
                                           fc_b2, est, est, out + (size_t)t * 256 * 64,
                                           xin_t, 256, 192, 64, 0);
    // LSTM step on x_in = concat(forward_x[t], est)
    wmma_gemm<4><<<gGate, blk, 0, stream>>>((const unsigned int*)xin_t, 256, Wih0_p,
                                            (const unsigned int*)h0, 1024, Whh0_p,
                                            bias0, nullptr, z, nullptr,
                                            nullptr, 0, 0, 4096, 0);
    lstm_cell<<<gCell, blk, 0, stream>>>(z, c0, h0);
    wmma_gemm<4><<<gGate, blk, 0, stream>>>((const unsigned int*)h0, 1024, Wih1_p,
                                            (const unsigned int*)h1, 1024, Whh1_p,
                                            bias1, nullptr, z, nullptr,
                                            nullptr, 0, 0, 4096, 0);
    lstm_cell<<<gCell, blk, 0, stream>>>(z, c1, h1);
  }
}
